// BidirectionalLaCTSwiGLU_6390911336716
// MI455X (gfx1250) — compile-verified, matrix operand order checked by reading, not run on hardware
//
#include <hip/hip_runtime.h>

// ------------------------------------------------------------------ types
typedef __bf16 bf16;
typedef __attribute__((ext_vector_type(16))) __bf16 v16bf;
typedef __attribute__((ext_vector_type(8)))  float  v8f;

#define STR   136  // LDS row stride (bf16) for 128-wide tiles: 272B = 17*16B (aligned, bank-skewed)
#define K2    64   // K panel depth for the big GEMMs
#define K2STR 72   // LDS row stride (bf16) for K=64 panels: 144B = 9*16B

#define SEQ    4096
#define DMODEL 2048
#define NQKV   6144
#define HD     128

// ------------------------------------------------------------------ helpers
static __device__ inline bf16 f2bf(float f) {
  union { float f; unsigned u; } x; x.f = f;
  unsigned r = x.u + 0x7FFFu + ((x.u >> 16) & 1u);
  union { unsigned short s; bf16 b; } y; y.s = (unsigned short)(r >> 16);
  return y.b;
}
static __device__ inline float bf2f(bf16 b) {
  union { unsigned short s; bf16 b; } y; y.b = b;
  union { unsigned u; float f; } x; x.u = ((unsigned)y.s) << 16;
  return x.f;
}
static __device__ inline unsigned pkbf(float a, float b) {
  union { float f; unsigned u; } xa; xa.f = a;
  union { float f; unsigned u; } xb; xb.f = b;
  unsigned ra = (xa.u + 0x7FFFu + ((xa.u >> 16) & 1u)) >> 16;
  unsigned rb = (xb.u + 0x7FFFu + ((xb.u >> 16) & 1u)) >> 16;
  return (rb << 16) | (ra & 0xFFFFu);
}
static __device__ inline float sigmf(float x) { return 1.f / (1.f + __expf(-x)); }

// Async global->LDS B128 copy (GVS mode: SGPR 64b base + per-lane 32b byte offset).
// LDS destination address = low 32 bits of the generic pointer.  Tracked by ASYNCcnt.
static __device__ inline void async_b128(void* lds, unsigned goff_bytes,
                                         unsigned long long gbase) {
  unsigned ldsa = (unsigned)(uintptr_t)lds;
  asm volatile("global_load_async_to_lds_b128 %0, %1, %2"
               :: "v"(ldsa), "v"(goff_bytes), "s"(gbase) : "memory");
}
static __device__ inline void async_drain() {
  asm volatile("s_wait_asynccnt 0x0" ::: "memory");
}

// Fragment load: lane (l&15) selects row (M for A-operand, N for B-operand),
// halves 0..7 cover K = (lane>>4)*8 .., halves 8..15 cover K+16 (ISA 7.12.2 layout).
// Two ds_load_b128 per fragment.
static __device__ inline v16bf frag_ld(const bf16* p, int stride) {
  int lane = threadIdx.x & 31;
  const bf16* r = p + (lane & 15) * stride + ((lane >> 4) << 3);
  union { v16bf v; uint4 q[2]; } f;
  f.q[0] = *(const uint4*)(r);
  f.q[1] = *(const uint4*)(r + 16);
  return f.v;
}

static __device__ inline v8f wmma_bf(v16bf a, v16bf b, v8f c) {
  return __builtin_amdgcn_wmma_f32_16x16x32_bf16(false, a, false, b, (short)0, c, false, false);
}

static __device__ inline void zero_acc(v8f C[2][4]) {
  v8f z = {0.f,0.f,0.f,0.f,0.f,0.f,0.f,0.f};
#pragma unroll
  for (int i = 0; i < 2; ++i)
#pragma unroll
    for (int j = 0; j < 4; ++j) C[i][j] = z;
}

// One K=32 WMMA step: wave computes its 32x64 patch (2x4 tiles of 16x16).
static __device__ inline void mm_step(const bf16* A, int sa, const bf16* B, int sb,
                                      v8f C[2][4], int pwm, int pwn) {
  const bf16* Ab = A + (pwm * 32) * sa;
  const bf16* Bb = B + (pwn * 64) * sb;
  v16bf a0 = frag_ld(Ab, sa);
  v16bf a1 = frag_ld(Ab + 16 * sa, sa);
#pragma unroll
  for (int ni = 0; ni < 4; ++ni) {
    v16bf b = frag_ld(Bb + ni * 16 * sb, sb);
    C[0][ni] = wmma_bf(a0, b, C[0][ni]);
    C[1][ni] = wmma_bf(a1, b, C[1][ni]);
  }
}

// Full K=128 matmul over LDS-resident 128x128 operands (stride STR).
static __device__ inline void mm128(const bf16* A, const bf16* B, v8f C[2][4],
                                    int pwm, int pwn) {
#pragma unroll
  for (int kk = 0; kk < 128; kk += 32)
    mm_step(A + kk, STR, B + kk, STR, C, pwm, pwn);
}

// Iterate the wave's accumulator elements; er/ec are row/col within the 128x128 tile.
template <typename F>
static __device__ inline void elem_loop(int lane, int pwm, int pwn, F f) {
#pragma unroll
  for (int mi = 0; mi < 2; ++mi)
#pragma unroll
    for (int ni = 0; ni < 4; ++ni)
#pragma unroll
      for (int e = 0; e < 8; ++e) {
        int er = pwm * 32 + mi * 16 + e + ((lane >> 4) << 3);
        int ec = pwn * 64 + ni * 16 + (lane & 15);
        f(mi, ni, e, er, ec);
      }
}

// ================================================================== K1: qkv = silu(x@Wqkv)
// grid (48, 64). Double-buffered K=64 panels via an even/odd unrolled loop with
// constant-offset LDS buffers (keeps InferAddressSpaces happy -> ds_load_b128).
__global__ __launch_bounds__(256) void k_qkv(const float* __restrict__ x,
                                             const float* __restrict__ Wqkv,
                                             bf16* __restrict__ qb,
                                             bf16* __restrict__ kb,
                                             bf16* __restrict__ vb) {
  extern __shared__ char smem[];
  bf16* A0  = (bf16*)smem;
  bf16* Bt0 = A0  + 128 * K2STR;
  bf16* A1  = Bt0 + 128 * K2STR;
  bf16* Bt1 = A1  + 128 * K2STR;
  float* red = (float*)(Bt1 + 128 * K2STR);
  int tid = threadIdx.x, lane = tid & 31, wave = tid >> 5;
  int pwm = wave & 3, pwn = wave >> 2;
  int nt = blockIdx.x;
  int m0 = blockIdx.y * 128, n0 = nt * 128;
  int which = nt >> 4, head = nt & 15;
  if (tid < 128) red[tid] = 0.f;
  v8f C[2][4]; zero_acc(C);

  auto load_tile = [&](bf16* Ad, bf16* Bd, int k0) {
    for (int i = tid; i < 128 * 16; i += 256) {           // A: 128 x 64 f32 -> bf16
      int r = i >> 4, q = (i & 15) << 2;
      float4 f = *(const float4*)(x + (size_t)(m0 + r) * DMODEL + k0 + q);
      uint2 p; p.x = pkbf(f.x, f.y); p.y = pkbf(f.z, f.w);
      *(uint2*)&Ad[r * K2STR + q] = p;
    }
    for (int i = tid; i < 64 * 32; i += 256) {            // B: [n][k] transposed store
      int kk = i >> 5, g = (i & 31) << 2;
      float4 f = *(const float4*)(Wqkv + (size_t)(k0 + kk) * NQKV + n0 + g);
      Bd[(g + 0) * K2STR + kk] = f2bf(f.x);
      Bd[(g + 1) * K2STR + kk] = f2bf(f.y);
      Bd[(g + 2) * K2STR + kk] = f2bf(f.z);
      Bd[(g + 3) * K2STR + kk] = f2bf(f.w);
    }
  };

  load_tile(A0, Bt0, 0);
  __syncthreads();
  for (int k0 = 0; k0 < DMODEL; k0 += 2 * K2) {           // DMODEL/K2 = 32 (even)
    load_tile(A1, Bt1, k0 + K2);                          // overlaps A0 compute
    if (k0 + 2 * K2 < DMODEL) {
      __builtin_prefetch(x + (size_t)(m0 + (tid >> 1)) * DMODEL + k0 + 2 * K2 + ((tid & 1) << 5), 0, 3);
      __builtin_prefetch(Wqkv + (size_t)(k0 + 2 * K2 + (tid >> 2)) * NQKV + n0 + ((tid & 3) << 5), 0, 3);
    }
    mm_step(A0,      K2STR, Bt0,      K2STR, C, pwm, pwn);
    mm_step(A0 + 32, K2STR, Bt0 + 32, K2STR, C, pwm, pwn);
    __syncthreads();
    if (k0 + 2 * K2 < DMODEL) load_tile(A0, Bt0, k0 + 2 * K2);   // overlaps A1 compute
    mm_step(A1,      K2STR, Bt1,      K2STR, C, pwm, pwn);
    mm_step(A1 + 32, K2STR, Bt1 + 32, K2STR, C, pwm, pwn);
    __syncthreads();
  }

  // silu, then L2-norm (q/k only; N tile == full head dim, so it's block-local)
  elem_loop(lane, pwm, pwn, [&](int mi, int ni, int e, int er, int ec) {
    (void)ec;
    float g = C[mi][ni][e];
    float s = g * sigmf(g);
    C[mi][ni][e] = s;
    if (which < 2) atomicAdd(&red[er], s * s);
  });
  __syncthreads();
  bf16* dst = which == 0 ? qb : (which == 1 ? kb : vb);
  elem_loop(lane, pwm, pwn, [&](int mi, int ni, int e, int er, int ec) {
    float s = C[mi][ni][e];
    if (which < 2) s *= 1.f / (sqrtf(red[er]) + 1e-5f);
    int m = m0 + er;
    int b = m >> 12, l = m & 4095;
    int bh = b * 16 + head;
    dst[((size_t)bh * SEQ + l) * HD + ec] = f2bf(s);
  });
}

// ================================================================== K2: lr = softplus(x@Wlr + c)
__global__ __launch_bounds__(256) void k_lr(const float* __restrict__ x,
                                            const float* __restrict__ Wlr,
                                            float* __restrict__ lrb) {
  int wave = threadIdx.x >> 5, lane = threadIdx.x & 31;
  int idx = blockIdx.x * 8 + wave;     // one output per wave
  int row = idx / 48, j = idx - row * 48;
  float s = 0.f;
  for (int k = lane; k < DMODEL; k += 32)
    s += x[(size_t)row * DMODEL + k] * Wlr[(size_t)k * 48 + j];
#pragma unroll
  for (int m = 16; m >= 1; m >>= 1) s += __shfl_xor(s, m, 32);
  if (lane == 0) {
    float t = s - 4.6001702f;                  // + inverse-softplus(0.01)
    float sp = t > 15.f ? t : log1pf(__expf(t));
    int which = j >> 4, hh = j & 15;
    int b = row >> 12, l = row & 4095;
    lrb[((size_t)which * 32 + (b * 16 + hh)) * SEQ + l] = sp;
  }
}

// ================================================================== K3: streaming dw accumulation
// grid (8 L-segments, 32 heads). Five 128x128 bf16 LDS buffers (≈170KB of the WGP's
// 320KB LDS) reused across forward / transpose / backward phases per 128-token chunk.
// k/v chunk tiles are streamed with GLOBAL_LOAD_ASYNC_TO_LDS_B128.
__global__ __launch_bounds__(256) void k_dw(const bf16* __restrict__ kbuf,
                                            const bf16* __restrict__ vbuf,
                                            const float* __restrict__ w0g,
                                            const float* __restrict__ w1g,
                                            const float* __restrict__ w2g,
                                            const float* __restrict__ lrb,
                                            float* __restrict__ dwb) {
  extern __shared__ char smem[];
  bf16* B0 = (bf16*)smem;               // k[l][c]        -> hid_lr1[i][l]
  bf16* B1 = B0 + 128 * STR;            // v[l][c]        -> dhbm_lr2[i][l]
  bf16* B2 = B1 + 128 * STR;            // w0[i][c]       -> kT[j][l]
  bf16* B3 = B2 + 128 * STR;            // w2[i][c]       -> vT[m][l]
  bf16* B4 = B3 + 128 * STR;            // w1T[i][c]      -> dgba_lr0[i][l]
  float* lr0s = (float*)(B4 + 128 * STR);
  float* lr1s = lr0s + 128;
  float* lr2s = lr1s + 128;
  int tid = threadIdx.x, lane = tid & 31, wave = tid >> 5;
  int pwm = wave & 3, pwn = wave >> 2;
  int seg = blockIdx.x, bh = blockIdx.y, hh = bh & 15;

  v8f D0[2][4], D1[2][4], D2[2][4];
  zero_acc(D0); zero_acc(D1); zero_acc(D2);

  for (int ch = 0; ch < 4; ++ch) {
    int l0 = seg * 512 + ch * 128;
    const bf16* ksrc = kbuf + ((size_t)bh * SEQ + l0) * HD;
    const bf16* vsrc = vbuf + ((size_t)bh * SEQ + l0) * HD;
    unsigned long long kbase = (unsigned long long)(uintptr_t)ksrc;
    unsigned long long vbase = (unsigned long long)(uintptr_t)vsrc;
    for (int i = tid; i < 128 * 16; i += 256) {          // async raw bf16 copies
      int r = i >> 4, q8 = (i & 15) * 8;
      unsigned vo = (unsigned)(((size_t)r * HD + q8) * sizeof(bf16));
      async_b128(&B0[r * STR + q8], vo, kbase);
      async_b128(&B1[r * STR + q8], vo, vbase);
    }
    for (int i = tid; i < 128 * 128; i += 256) {         // f32 -> bf16 (VALU path)
      int r = i >> 7, c = i & 127;
      B2[r * STR + c] = f2bf(w0g[((size_t)hh * HD + r) * HD + c]);
      B3[r * STR + c] = f2bf(w2g[((size_t)hh * HD + r) * HD + c]);
      B4[r * STR + c] = f2bf(w1g[((size_t)hh * HD + c) * HD + r]);   // w1^T
    }
    if (tid < 128) {
      lr0s[tid] = lrb[((size_t)0 * 32 + bh) * SEQ + l0 + tid];
      lr1s[tid] = lrb[((size_t)1 * 32 + bh) * SEQ + l0 + tid];
      lr2s[tid] = lrb[((size_t)2 * 32 + bh) * SEQ + l0 + tid];
    }
    if (ch + 1 < 4) {   // L2 prefetch of next chunk's k/v rows
      size_t po = (size_t)(tid & 127) * HD + ((tid >> 7) << 6);
      __builtin_prefetch(ksrc + (size_t)128 * HD + po, 0, 3);
      __builtin_prefetch(vsrc + (size_t)128 * HD + po, 0, 3);
    }
    async_drain();
    __syncthreads();

    // Forward (transposed): gateT = k@w0^T, hbT = k@w2^T, dhT = v@w1
    v8f Cg[2][4], Chb[2][4], Cd[2][4];
    zero_acc(Cg); zero_acc(Chb); zero_acc(Cd);
    mm128(B0, B2, Cg,  pwm, pwn);
    mm128(B0, B3, Chb, pwm, pwn);
    mm128(B1, B4, Cd,  pwm, pwn);
    __syncthreads();

    // Phase A: kT->B2, vT->B3, dgba*lr0 -> B4 (reads B0/B1/regs only)
    for (int i = tid; i < 128 * 128; i += 256) {
      int r = i >> 7, c = i & 127;
      B2[r * STR + c] = B0[c * STR + r];
      B3[r * STR + c] = B1[c * STR + r];
    }
    elem_loop(lane, pwm, pwn, [&](int mi, int ni, int e, int er, int ec) {
      float g = Cg[mi][ni][e], hb = Chb[mi][ni][e], d = Cd[mi][ni][e];
      float sg = sigmf(g);
      float dgba = (d * hb) * (sg * (1.f + g * (1.f - sg)));
      B4[ec * STR + er] = f2bf(dgba * lr0s[er]);
    });
    __syncthreads();

    // Phase B: hid*lr1 -> B0, dhbm*lr2 -> B1 (from regs)
    elem_loop(lane, pwm, pwn, [&](int mi, int ni, int e, int er, int ec) {
      float g = Cg[mi][ni][e], hb = Chb[mi][ni][e], d = Cd[mi][ni][e];
      float sil = g * sigmf(g);
      B0[ec * STR + er] = f2bf(sil * hb * lr1s[er]);
      B1[ec * STR + er] = f2bf(d * sil * lr2s[er]);
    });
    __syncthreads();

    // Backward: dw0 += dgba_lr0 @ k ; dw2 += dhbm_lr2 @ k ; dw1 += vT @ (hid*lr1)^T
    mm128(B4, B2, D0, pwm, pwn);
    mm128(B1, B2, D2, pwm, pwn);
    mm128(B3, B0, D1, pwm, pwn);
    __syncthreads();
  }

  float* d0 = dwb + ((size_t)0 * 32 + bh) * (HD * HD);
  float* d1 = dwb + ((size_t)1 * 32 + bh) * (HD * HD);
  float* d2 = dwb + ((size_t)2 * 32 + bh) * (HD * HD);
  elem_loop(lane, pwm, pwn, [&](int mi, int ni, int e, int er, int ec) {
    atomicAdd(&d0[er * HD + ec], D0[mi][ni][e]);
    atomicAdd(&d1[er * HD + ec], D1[mi][ni][e]);
    atomicAdd(&d2[er * HD + ec], D2[mi][ni][e]);
  });
}

// ================================================================== K4: NS5 + weight update + renorm
// grid (96) = 3 mats x 32 heads. X, XT, A, B (bf16) + f32 gradient all LDS-resident.
// Gradient is streamed in with async-to-LDS, then reduced from LDS.
__global__ __launch_bounds__(256) void k_ns5(const float* __restrict__ dwb,
                                             const float* __restrict__ w0g,
                                             const float* __restrict__ w1g,
                                             const float* __restrict__ w2g,
                                             bf16* __restrict__ wub) {
  extern __shared__ char smem[];
  bf16* X  = (bf16*)smem;
  bf16* XT = X  + 128 * STR;
  bf16* Am = XT + 128 * STR;
  bf16* Bm = Am + 128 * STR;
  float* Gs = (float*)(Bm + 128 * STR);
  float* tnorm = Gs + 128 * STR;
  float* nnorm = tnorm + 128;
  float* fn    = nnorm + 128;
  int tid = threadIdx.x, lane = tid & 31, wave = tid >> 5;
  int pwm = wave & 3, pwn = wave >> 2;
  int mat = blockIdx.x >> 5, bh = blockIdx.x & 31, hh = bh & 15;
  const float* G = dwb + ((size_t)mat * 32 + bh) * (HD * HD);
  const float* wsel = mat == 0 ? w0g : (mat == 1 ? w1g : w2g);
  unsigned long long gbase = (unsigned long long)(uintptr_t)G;

  if (tid == 0) fn[0] = 0.f;
  if (tid < 128) { tnorm[tid] = 0.f; nnorm[tid] = 0.f; }
  for (int i = tid; i < 128 * 32; i += 256) {     // async f32 copy: 32 x B128 per row
    int r = i >> 5, q4 = (i & 31) << 2;
    async_b128(&Gs[r * STR + q4], (unsigned)(((size_t)r * HD + q4) * sizeof(float)), gbase);
  }
  async_drain();
  __syncthreads();
  float part = 0.f;
  for (int i = tid; i < HD * HD; i += 256) {
    int r = i >> 7, c = i & 127;
    float g = Gs[r * STR + c];
    part += g * g;
  }
  atomicAdd(fn, part);
  __syncthreads();
  float sc = 1.f / (sqrtf(fn[0]) + 1e-7f);
  for (int i = tid; i < HD * HD; i += 256) {
    int r = i >> 7, c = i & 127;
    bf16 bv = f2bf(Gs[r * STR + c] * sc);
    X[r * STR + c] = bv; XT[c * STR + r] = bv;
  }
  __syncthreads();

  const float CA[5] = {4.0848f, 3.9505f, 3.7418f, 2.8769f, 2.8366f};
  const float CB[5] = {-6.8946f, -6.3029f, -5.5913f, -3.1427f, -3.0525f};
  const float CC[5] = {2.927f, 2.6377f, 2.3037f, 1.2046f, 1.2012f};
  for (int it = 0; it < 5; ++it) {
    v8f C1[2][4]; zero_acc(C1);
    mm128(X, X, C1, pwm, pwn);                      // A = X X^T (row-major X is its own B-op)
    elem_loop(lane, pwm, pwn, [&](int mi, int ni, int e, int er, int ec) {
      Am[er * STR + ec] = f2bf(C1[mi][ni][e]);
    });
    __syncthreads();
    v8f C2[2][4]; zero_acc(C2);
    mm128(Am, Am, C2, pwm, pwn);                    // A@A (A symmetric)
    elem_loop(lane, pwm, pwn, [&](int mi, int ni, int e, int er, int ec) {
      Bm[er * STR + ec] = f2bf(CB[it] * bf2f(Am[er * STR + ec]) + CC[it] * C2[mi][ni][e]);
    });
    __syncthreads();
    v8f C3[2][4]; zero_acc(C3);
    mm128(Bm, XT, C3, pwm, pwn);                    // B@X
    elem_loop(lane, pwm, pwn, [&](int mi, int ni, int e, int er, int ec) {
      C3[mi][ni][e] = CA[it] * bf2f(X[er * STR + ec]) + C3[mi][ni][e];
    });
    __syncthreads();
    elem_loop(lane, pwm, pwn, [&](int mi, int ni, int e, int er, int ec) {
      bf16 bv = f2bf(C3[mi][ni][e]);
      X[er * STR + ec] = bv; XT[ec * STR + er] = bv;
    });
    __syncthreads();
  }

  // w_new = w_old + ns5(dw) + dw ; renorm rows back to old row norms
  for (int i = tid; i < HD * HD; i += 256) {
    int r = i >> 7, c = i & 127;
    float wo = wsel[((size_t)hh * HD + r) * HD + c];
    float wn = wo + bf2f(X[r * STR + c]) + Gs[r * STR + c];
    Gs[r * STR + c] = wn;
    atomicAdd(&tnorm[r], wo * wo);
    atomicAdd(&nnorm[r], wn * wn);
  }
  __syncthreads();
  bf16* dst = wub + ((size_t)mat * 32 + bh) * (HD * HD);
  for (int i = tid; i < HD * HD; i += 256) {
    int r = i >> 7, c = i & 127;
    float scale = sqrtf(tnorm[r]) / (sqrtf(nnorm[r]) + 1e-5f);
    dst[i] = f2bf(Gs[r * STR + c] * scale);
  }
}

// ================================================================== K5: apply + RMSNorm
// grid (32 L-tiles, 32 heads). oT = silu(q@w0^T)*(q@w2^T) @ w1^T, then rms-norm rows.
// All four input tiles are raw bf16 copies -> async-to-LDS.
__global__ __launch_bounds__(256) void k_apply(const bf16* __restrict__ qbuf,
                                               const bf16* __restrict__ wub,
                                               const float* __restrict__ osc,
                                               bf16* __restrict__ ob) {
  extern __shared__ char smem[];
  bf16* Q   = (bf16*)smem;
  bf16* W0s = Q   + 128 * STR;
  bf16* W2s = W0s + 128 * STR;
  bf16* W1s = W2s + 128 * STR;
  bf16* GH  = W1s + 128 * STR;
  float* osum = (float*)(GH + 128 * STR);
  float* oscs = osum + 128;
  int tid = threadIdx.x, lane = tid & 31, wave = tid >> 5;
  int pwm = wave & 3, pwn = wave >> 2;
  int lt = blockIdx.x, bh = blockIdx.y;
  int b = bh >> 4, hh = bh & 15;
  int l0 = lt * 128;

  const bf16* qsrc = qbuf + ((size_t)bh * SEQ + l0) * HD;
  const bf16* w0u = wub + ((size_t)0 * 32 + bh) * (HD * HD);
  const bf16* w1u = wub + ((size_t)1 * 32 + bh) * (HD * HD);
  const bf16* w2u = wub + ((size_t)2 * 32 + bh) * (HD * HD);
  {
    unsigned long long qa = (unsigned long long)(uintptr_t)qsrc;
    unsigned long long a0 = (unsigned long long)(uintptr_t)w0u;
    unsigned long long a1 = (unsigned long long)(uintptr_t)w1u;
    unsigned long long a2 = (unsigned long long)(uintptr_t)w2u;
    for (int i = tid; i < 128 * 16; i += 256) {
      int r = i >> 4, q8 = (i & 15) * 8;
      unsigned vo = (unsigned)(((size_t)r * HD + q8) * sizeof(bf16));
      async_b128(&Q[r * STR + q8],   vo, qa);
      async_b128(&W0s[r * STR + q8], vo, a0);
      async_b128(&W2s[r * STR + q8], vo, a2);
      async_b128(&W1s[r * STR + q8], vo, a1);
    }
  }
  if (tid < 128) { osum[tid] = 0.f; oscs[tid] = osc[tid]; }
  async_drain();
  __syncthreads();

  v8f Cg[2][4], Chh[2][4];
  zero_acc(Cg); zero_acc(Chh);
  mm128(Q, W0s, Cg,  pwm, pwn);     // gateT = q @ w0^T
  mm128(Q, W2s, Chh, pwm, pwn);     // hT    = q @ w2^T
  elem_loop(lane, pwm, pwn, [&](int mi, int ni, int e, int er, int ec) {
    float g = Cg[mi][ni][e];
    GH[er * STR + ec] = f2bf(g * sigmf(g) * Chh[mi][ni][e]);
  });
  __syncthreads();
  v8f Co[2][4]; zero_acc(Co);
  mm128(GH, W1s, Co, pwm, pwn);     // oT = (gate*h)T @ w1^T
  elem_loop(lane, pwm, pwn, [&](int mi, int ni, int e, int er, int ec) {
    (void)ec;
    float v = Co[mi][ni][e];
    atomicAdd(&osum[er], v * v);
  });
  __syncthreads();
  elem_loop(lane, pwm, pwn, [&](int mi, int ni, int e, int er, int ec) {
    float norm = sqrtf(osum[er] * (1.f / 128.f) + 1e-5f);
    float v = oscs[ec] * Co[mi][ni][e] / norm;
    ob[(((size_t)(b * SEQ + l0 + er)) * 16 + hh) * HD + ec] = f2bf(v);
  });
}

// ================================================================== K6: out = o_n @ Wo (f32 out)
// Even/odd-unrolled double buffering (constant LDS pointers); A operand streamed with
// async-to-LDS behind the WMMAs, drained just before each publishing barrier.
__global__ __launch_bounds__(256) void k_out(const bf16* __restrict__ ob,
                                             const float* __restrict__ Wo,
                                             float* __restrict__ out) {
  extern __shared__ char smem[];
  bf16* A0  = (bf16*)smem;
  bf16* Bt0 = A0  + 128 * K2STR;
  bf16* A1  = Bt0 + 128 * K2STR;
  bf16* Bt1 = A1  + 128 * K2STR;
  int tid = threadIdx.x, lane = tid & 31, wave = tid >> 5;
  int pwm = wave & 3, pwn = wave >> 2;
  int n0 = blockIdx.x * 128, m0 = blockIdx.y * 128;
  unsigned long long obase = (unsigned long long)(uintptr_t)ob;

  auto load_A_async = [&](bf16* Ad, int k0) {
    for (int i = tid; i < 128 * 8; i += 256) {       // 8 x B128 per row of 64 halves
      int r = i >> 3, q = (i & 7) << 3;
      async_b128(&Ad[r * K2STR + q],
                 (unsigned)(((size_t)(m0 + r) * DMODEL + k0 + q) * sizeof(bf16)), obase);
    }
  };
  auto load_B = [&](bf16* Bd, int k0) {
    for (int i = tid; i < 64 * 32; i += 256) {
      int kk = i >> 5, g = (i & 31) << 2;
      float4 f = *(const float4*)(Wo + (size_t)(k0 + kk) * DMODEL + n0 + g);
      Bd[(g + 0) * K2STR + kk] = f2bf(f.x);
      Bd[(g + 1) * K2STR + kk] = f2bf(f.y);
      Bd[(g + 2) * K2STR + kk] = f2bf(f.z);
      Bd[(g + 3) * K2STR + kk] = f2bf(f.w);
    }
  };

  v8f C[2][4]; zero_acc(C);
  load_A_async(A0, 0);
  load_B(Bt0, 0);
  async_drain();
  __syncthreads();
  for (int k0 = 0; k0 < DMODEL; k0 += 2 * K2) {
    load_A_async(A1, k0 + K2);                       // behind the WMMAs below
    load_B(Bt1, k0 + K2);
    if (k0 + 2 * K2 < DMODEL)
      __builtin_prefetch(Wo + (size_t)(k0 + 2 * K2 + (tid >> 2)) * DMODEL + n0 + ((tid & 3) << 5), 0, 3);
    mm_step(A0,      K2STR, Bt0,      K2STR, C, pwm, pwn);
    mm_step(A0 + 32, K2STR, Bt0 + 32, K2STR, C, pwm, pwn);
    async_drain();
    __syncthreads();
    if (k0 + 2 * K2 < DMODEL) {
      load_A_async(A0, k0 + 2 * K2);
      load_B(Bt0, k0 + 2 * K2);
    }
    mm_step(A1,      K2STR, Bt1,      K2STR, C, pwm, pwn);
    mm_step(A1 + 32, K2STR, Bt1 + 32, K2STR, C, pwm, pwn);
    async_drain();
    __syncthreads();
  }
  elem_loop(lane, pwm, pwn, [&](int mi, int ni, int e, int er, int ec) {
    out[(size_t)(m0 + er) * DMODEL + n0 + ec] = C[mi][ni][e];
  });
}

// ================================================================== host
extern "C" void kernel_launch(void* const* d_in, const int* in_sizes, int n_in,
                              void* d_out, int out_size, void* d_ws, size_t ws_size,
                              hipStream_t stream) {
  (void)in_sizes; (void)n_in; (void)out_size; (void)ws_size;
  const float* x    = (const float*)d_in[0];
  const float* Wqkv = (const float*)d_in[1];
  const float* Wo   = (const float*)d_in[2];
  const float* Wlr  = (const float*)d_in[3];
  const float* w0   = (const float*)d_in[4];
  const float* w1   = (const float*)d_in[5];
  const float* w2   = (const float*)d_in[6];
  const float* osc  = (const float*)d_in[7];
  float* out = (float*)d_out;

  char* ws = (char*)d_ws;
  const size_t MB = 1024 * 1024;
  bf16*  qb  = (bf16*)(ws + 0 * MB);      // 32MB : q (silu+l2norm), bf16 [BH][L][hd]
  bf16*  kb  = (bf16*)(ws + 32 * MB);     // 32MB : k
  bf16*  vb  = (bf16*)(ws + 64 * MB);     // 32MB : v
  float* lrb = (float*)(ws + 96 * MB);    // 1.5MB: lr [3][BH][L]
  float* dwb = (float*)(ws + 98 * MB);    // 6MB  : dw [3][BH][128][128] f32
  bf16*  wub = (bf16*)(ws + 105 * MB);    // 3MB  : updated weights bf16
  bf16*  ob  = (bf16*)(ws + 109 * MB);    // 32MB : o normalized bf16 [B][L][H][hd]

  size_t s1 = (size_t)4 * 128 * K2STR * sizeof(bf16) + 128 * sizeof(float);
  k_qkv<<<dim3(48, 64), 256, s1, stream>>>(x, Wqkv, qb, kb, vb);
  k_lr<<<dim3(49152), 256, 0, stream>>>(x, Wlr, lrb);
  hipMemsetAsync(dwb, 0, (size_t)3 * 32 * HD * HD * sizeof(float), stream);
  size_t s3 = (size_t)5 * 128 * STR * sizeof(bf16) + 3 * 128 * sizeof(float);
  k_dw<<<dim3(8, 32), 256, s3, stream>>>(kb, vb, w0, w1, w2, lrb, dwb);
  size_t s4 = (size_t)4 * 128 * STR * sizeof(bf16) + (size_t)128 * STR * sizeof(float)
            + (2 * 128 + 4) * sizeof(float);
  k_ns5<<<dim3(96), 256, s4, stream>>>(dwb, w0, w1, w2, wub);
  size_t s5 = (size_t)5 * 128 * STR * sizeof(bf16) + 2 * 128 * sizeof(float);
  k_apply<<<dim3(32, 32), 256, s5, stream>>>(qb, wub, osc, ob);
  size_t s6 = (size_t)4 * 128 * K2STR * sizeof(bf16);
  k_out<<<dim3(16, 64), 256, s6, stream>>>(ob, Wo, out);
}
